// A2C_65455301591869
// MI455X (gfx1250) — compile-verified, matrix operand order checked by reading, not run on hardware
//
#include <hip/hip_runtime.h>
#include <hip/hip_bf16.h>
#include <math.h>

typedef __attribute__((ext_vector_type(16))) _Float16 v16h;
typedef __attribute__((ext_vector_type(8)))  float    v8f;

union FragH {
    v16h v;
    _Float16 h[16];
    unsigned int u[8];
};

union Pack4 {
    _Float16 h[4];
    uint2 u2;
};

__device__ __forceinline__ float softplus_f(float x) {
    return (x > 20.0f) ? x : log1pf(expf(x));
}

// float atomic max via int punning (valid with -inf initialization)
__device__ __forceinline__ void atomicMaxF(float* addr, float val) {
    if (val >= 0.0f) atomicMax((int*)addr, __float_as_int(val));
    else             atomicMin((unsigned int*)addr, __float_as_uint(val));
}

// ---------------------------------------------------------------------------
// elementwise helpers
// ---------------------------------------------------------------------------
__global__ void fill_kernel(float* __restrict__ p, float v, long n) {
    long i = (long)blockIdx.x * blockDim.x + threadIdx.x;
    long s = (long)gridDim.x * blockDim.x;
    for (; i < n; i += s) p[i] = v;
}

__global__ void deg_kernel(const int* __restrict__ dst, float* __restrict__ deg, int E) {
    int i = blockIdx.x * blockDim.x + threadIdx.x;
    int s = gridDim.x * blockDim.x;
    for (; i < E; i += s) atomicAdd(&deg[dst[i]], 1.0f);
}

__global__ void rsqrt_kernel(float* __restrict__ d, int n) {
    int i = blockIdx.x * blockDim.x + threadIdx.x;
    if (i < n) d[i] = rsqrtf(d[i]);
}

// acc[i,f] = hw[i,f] * dis[i]^2   (self-loop term, also zero-inits acc)
__global__ void gcn_init_kernel(const float* __restrict__ hw, const float* __restrict__ dis,
                                float* __restrict__ acc, long n, int hshift) {
    long i = (long)blockIdx.x * blockDim.x + threadIdx.x;
    long s = (long)gridDim.x * blockDim.x;
    for (; i < n; i += s) {
        int node = (int)(i >> hshift);
        float d = dis[node];
        acc[i] = hw[i] * d * d;
    }
}

// per edge: acc[dst, :] += hw[src, :] * dis[src]*dis[dst]
__global__ void gcn_scatter_kernel(const float* __restrict__ hw,
                                   const int* __restrict__ src, const int* __restrict__ dst,
                                   const float* __restrict__ dis, float* __restrict__ acc,
                                   int E, int H) {
    int e = blockIdx.x;
    if (e >= E) return;
    int s = src[e], d = dst[e];
    float norm = dis[s] * dis[d];
    const float* hs = hw + (long)s * H;
    float* ad = acc + (long)d * H;
    for (int f = threadIdx.x; f < H; f += blockDim.x)
        atomicAdd(&ad[f], hs[f] * norm);
}

__global__ void bias_act_kernel(const float* __restrict__ acc, const float* __restrict__ bias,
                                float* __restrict__ out, long n, int mask, int act) {
    long i = (long)blockIdx.x * blockDim.x + threadIdx.x;
    long s = (long)gridDim.x * blockDim.x;
    for (; i < n; i += s) {
        float v = acc[i] + bias[i & mask];
        if (act == 1) v = fmaxf(v, 0.0f);
        else if (act == 2) v = softplus_f(v);
        out[i] = v;
    }
}

__global__ void agg_fin_kernel(float* __restrict__ agg, long n) {
    long i = (long)blockIdx.x * blockDim.x + threadIdx.x;
    long s = (long)gridDim.x * blockDim.x;
    for (; i < n; i += s) {
        float v = agg[i];
        if (!(fabsf(v) < 3.0e38f)) v = 0.0f;   // -inf / nan -> 0
        agg[i] = fmaxf(v, 0.0f);               // relu
    }
}

// C[row, 0:wa] = A[row], C[row, wa:wa+wb] = B[row];  wa+wb == 1<<wshift
__global__ void concat_kernel(const float* __restrict__ A, int wa,
                              const float* __restrict__ B, int wb,
                              float* __restrict__ C, long n, int wshift) {
    long i = (long)blockIdx.x * blockDim.x + threadIdx.x;
    long s = (long)gridDim.x * blockDim.x;
    int wm = (1 << wshift) - 1;
    for (; i < n; i += s) {
        long row = i >> wshift;
        int c = (int)(i & wm);
        C[i] = (c < wa) ? A[row * wa + c] : B[row * wb + (c - wa)];
    }
}

// [N,32] @ [32,2] + b -> d_out[0:N]=col0, d_out[N:2N]=col1
__global__ void final_kernel(const float* __restrict__ h, const float* __restrict__ W,
                             const float* __restrict__ b, float* __restrict__ out, int N) {
    int i = blockIdx.x * blockDim.x + threadIdx.x;
    if (i >= N) return;
    float o0 = b[0], o1 = b[1];
    const float* hi = h + (long)i * 32;
    #pragma unroll
    for (int k = 0; k < 32; ++k) {
        float v = hi[k];
        o0 += v * W[k * 2 + 0];
        o1 += v * W[k * 2 + 1];
    }
    out[i] = o0;
    out[N + i] = o1;
}

// ---------------------------------------------------------------------------
// Weight repack: W[K,Ncol] f32 -> f16 WMMA-B fragment layout.
// Linear index: ((ct*ksteps + ks)*32 + lane)*16 + j
// j = 2v+e;  kk = (v>>2)*16 + (lane>>4)*8 + (v&3)*2 + e;  col = ct*16 + (lane&15)
// ---------------------------------------------------------------------------
__global__ void repack_kernel(const float* __restrict__ W, _Float16* __restrict__ out,
                              int K, int Ncol) {
    int ksteps = K >> 5;
    long total = (long)(Ncol >> 4) * ksteps * 512;
    long i = (long)blockIdx.x * blockDim.x + threadIdx.x;
    long s = (long)gridDim.x * blockDim.x;
    for (; i < total; i += s) {
        int per_ct = ksteps * 512;
        int ct = (int)(i / per_ct);
        int rem = (int)(i - (long)ct * per_ct);
        int ks = rem >> 9;
        int r2 = rem & 511;
        int lane = r2 >> 4;
        int j = r2 & 15;
        int v = j >> 1, e = j & 1;
        int kk = ((v >> 2) << 4) + ((lane >> 4) << 3) + ((v & 3) << 1) + e;
        int k = (ks << 5) + kk;
        int col = (ct << 4) + (lane & 15);
        out[i] = (_Float16)W[(long)k * Ncol + col];
    }
}

// ---------------------------------------------------------------------------
// WMMA GEMM v2: C[M, Ncol] = act(A[M,K] @ W + bias)
// Block: 256 threads = 8 waves, covers 32 rows x 128 cols (grid.y tiles cols).
// Full 32 x K f16 A-tile staged once in dynamic LDS (float4 loads), then a
// clean k-loop: 2 A-frag b128 LDS loads + 1 B-frag 32B load + 2 wmma per step
// (two row tiles share one B fragment -> 2 independent accumulator chains).
// Requires M % 32 == 0, K % 32 == 0 (K <= 512), Ncol % 16 == 0.
// ---------------------------------------------------------------------------
__global__ void __launch_bounds__(256)
wmma_gemm_kernel(const float* __restrict__ A, int lda,
                 const _Float16* __restrict__ Wp, const float* __restrict__ bias,
                 float* __restrict__ C, int ldc, int K, int Ncol, int act) {
    extern __shared__ _Float16 ldsA[];   // 32 * K halves
    const int t = threadIdx.x;
    const int wave = t >> 5;
    const int lane = t & 31;
    const int row0 = blockIdx.x << 5;
    const int ct = (blockIdx.y << 3) + wave;
    const int ksteps = K >> 5;
    const int nct = Ncol >> 4;
    const bool active = ct < nct;
    const int khalf = lane >> 4;
    const int colL = lane & 15;

    // ---- stage full 32 x K tile as f16 (vectorized) ----
    const int kq = K >> 2;                     // float4 per row (power of 2)
    const int kqsh = __builtin_ctz(kq);
    const int total4 = kq << 5;                // 32 rows
    for (int i = t; i < total4; i += 256) {
        int r = i >> kqsh;
        int q = i & (kq - 1);
        float4 fv = *(const float4*)(A + (long)(row0 + r) * lda + (q << 2));
        Pack4 p;
        p.h[0] = (_Float16)fv.x; p.h[1] = (_Float16)fv.y;
        p.h[2] = (_Float16)fv.z; p.h[3] = (_Float16)fv.w;
        *(uint2*)&ldsA[(long)r * K + (q << 2)] = p.u2;
    }
    __syncthreads();

    if (!active) return;

    const _Float16* wbase = Wp + (long)ct * ksteps * 512;
    const unsigned int* pa = (const unsigned int*)ldsA;
    const int rdw = K >> 1;                    // dwords per LDS row
    const int rowu0 = colL * rdw;              // row tile 0, row = colL
    const int rowu1 = (colL + 16) * rdw;       // row tile 1

    v8f acc0 = {}, acc1 = {};
    for (int ks = 0; ks < ksteps; ++ks) {
        FragH a0, a1, b;
        #pragma unroll
        for (int v = 0; v < 8; ++v) {
            int kw = ((v >> 2) << 3) + (khalf << 2) + (v & 3);
            a0.u[v] = pa[rowu0 + (ks << 4) + kw];
            a1.u[v] = pa[rowu1 + (ks << 4) + kw];
        }
        b.v = *(const v16h*)(wbase + ((long)(ks << 5) + lane) * 16);
        acc0 = __builtin_amdgcn_wmma_f32_16x16x32_f16(false, a0.v, false, b.v,
                                                      (short)0, acc0, false, false);
        acc1 = __builtin_amdgcn_wmma_f32_16x16x32_f16(false, a1.v, false, b.v,
                                                      (short)0, acc1, false, false);
    }

    const int col = (ct << 4) + colL;
    const float bb = bias ? bias[col] : 0.0f;
    #pragma unroll
    for (int r = 0; r < 8; ++r) {
        float v0 = acc0[r] + bb;
        float v1 = acc1[r] + bb;
        if (act == 1)      { v0 = fmaxf(v0, 0.0f);   v1 = fmaxf(v1, 0.0f); }
        else if (act == 2) { v0 = softplus_f(v0);    v1 = softplus_f(v1); }
        C[(long)(row0 + (khalf << 3) + r) * ldc + col] = v0;
        C[(long)(row0 + 16 + (khalf << 3) + r) * ldc + col] = v1;
    }
}

// ---------------------------------------------------------------------------
// Fused EdgeConv: per block = 16 edges.
//   m  = [x[dst] | x[src] | edge_attr]            (16 x 192, f16 in LDS)
//   h  = relu(m @ W1 + b1)                        (16 x 256, f16 in LDS)
//   o  = h @ W2 + b2                              (16 x 256, f32)
//   agg[dst] = atomic max(o)
// Each wave owns 2 col tiles; per k-step one A fragment feeds both wmmas.
// ---------------------------------------------------------------------------
__global__ void __launch_bounds__(256)
edgeconv_kernel(const float* __restrict__ x, const float* __restrict__ eattr,
                const int* __restrict__ src, const int* __restrict__ dst,
                const _Float16* __restrict__ W1p, const float* __restrict__ b1,
                const _Float16* __restrict__ W2p, const float* __restrict__ b2,
                float* __restrict__ agg, int E) {
    __shared__ _Float16 ldsIn[16 * 192];
    __shared__ _Float16 ldsHid[16 * 256];
    __shared__ int ldsSrc[16], ldsDst[16];

    const int t = threadIdx.x;
    const int wave = t >> 5;
    const int lane = t & 31;
    const int khalf = lane >> 4;
    const int colL = lane & 15;
    const int e0 = blockIdx.x << 4;

    if (t < 16) {
        int e = e0 + t;
        if (e >= E) e = E - 1;
        ldsSrc[t] = src[e];
        ldsDst[t] = dst[e];
    }
    __syncthreads();

    // gather 16 x 192 input tile: 16 edges x 3 segments x 16 float4 = 768 vec4
    for (int i = t; i < 768; i += 256) {
        int el = i / 48;                 // edge row 0..15
        int rem = i - el * 48;
        int seg = rem >> 4;              // 0: x[dst], 1: x[src], 2: eattr
        int q = rem & 15;                // float4 within 64-float row
        int er = e0 + el;
        if (er >= E) er = E - 1;
        const float* base;
        if (seg == 0)      base = x + (long)ldsDst[el] * 64;
        else if (seg == 1) base = x + (long)ldsSrc[el] * 64;
        else               base = eattr + (long)er * 64;
        float4 fv = *(const float4*)(base + (q << 2));
        Pack4 p;
        p.h[0] = (_Float16)fv.x; p.h[1] = (_Float16)fv.y;
        p.h[2] = (_Float16)fv.z; p.h[3] = (_Float16)fv.w;
        *(uint2*)&ldsIn[el * 192 + seg * 64 + (q << 2)] = p.u2;
    }
    __syncthreads();

    // ---- layer 1: K = 192 (6 k-steps), 2 col tiles per wave, shared A frag ----
    {
        const int ct0 = wave << 1;
        const _Float16* wb0 = W1p + (long)ct0 * 6 * 512;
        const _Float16* wb1 = wb0 + 6 * 512;
        const unsigned int* pa = (const unsigned int*)ldsIn;
        const int rowu = colL * 96;      // 192 halves = 96 dwords per row
        v8f acc0 = {}, acc1 = {};
        #pragma unroll
        for (int ks = 0; ks < 6; ++ks) {
            FragH a, bb0, bb1;
            #pragma unroll
            for (int v = 0; v < 8; ++v) {
                int kw = ((v >> 2) << 3) + (khalf << 2) + (v & 3);
                a.u[v] = pa[rowu + (ks << 4) + kw];
            }
            bb0.v = *(const v16h*)(wb0 + ((ks << 5) + lane) * 16);
            bb1.v = *(const v16h*)(wb1 + ((ks << 5) + lane) * 16);
            acc0 = __builtin_amdgcn_wmma_f32_16x16x32_f16(false, a.v, false, bb0.v,
                                                          (short)0, acc0, false, false);
            acc1 = __builtin_amdgcn_wmma_f32_16x16x32_f16(false, a.v, false, bb1.v,
                                                          (short)0, acc1, false, false);
        }
        int col0 = (ct0 << 4) + colL;
        float bv0 = b1[col0], bv1 = b1[col0 + 16];
        #pragma unroll
        for (int r = 0; r < 8; ++r) {
            int row = (khalf << 3) + r;
            ldsHid[row * 256 + col0]      = (_Float16)fmaxf(acc0[r] + bv0, 0.0f);
            ldsHid[row * 256 + col0 + 16] = (_Float16)fmaxf(acc1[r] + bv1, 0.0f);
        }
    }
    __syncthreads();

    // ---- layer 2: K = 256 (8 k-steps) + scatter-max aggregation ----
    {
        const int ct0 = wave << 1;
        const _Float16* wb0 = W2p + (long)ct0 * 8 * 512;
        const _Float16* wb1 = wb0 + 8 * 512;
        const unsigned int* pa = (const unsigned int*)ldsHid;
        const int rowu = colL << 7;      // 256 halves = 128 dwords per row
        v8f acc0 = {}, acc1 = {};
        #pragma unroll
        for (int ks = 0; ks < 8; ++ks) {
            FragH a, bb0, bb1;
            #pragma unroll
            for (int v = 0; v < 8; ++v) {
                int kw = ((v >> 2) << 3) + (khalf << 2) + (v & 3);
                a.u[v] = pa[rowu + (ks << 4) + kw];
            }
            bb0.v = *(const v16h*)(wb0 + ((ks << 5) + lane) * 16);
            bb1.v = *(const v16h*)(wb1 + ((ks << 5) + lane) * 16);
            acc0 = __builtin_amdgcn_wmma_f32_16x16x32_f16(false, a.v, false, bb0.v,
                                                          (short)0, acc0, false, false);
            acc1 = __builtin_amdgcn_wmma_f32_16x16x32_f16(false, a.v, false, bb1.v,
                                                          (short)0, acc1, false, false);
        }
        int col0 = (ct0 << 4) + colL;
        float bv0 = b2[col0], bv1 = b2[col0 + 16];
        #pragma unroll
        for (int r = 0; r < 8; ++r) {
            int row = (khalf << 3) + r;
            if (e0 + row < E) {
                float* ar = agg + (long)ldsDst[row] * 256;
                atomicMaxF(ar + col0,      acc0[r] + bv0);
                atomicMaxF(ar + col0 + 16, acc1[r] + bv1);
            }
        }
    }
}

// ---------------------------------------------------------------------------
extern "C" void kernel_launch(void* const* d_in, const int* in_sizes, int n_in,
                              void* d_out, int out_size, void* d_ws, size_t ws_size,
                              hipStream_t stream) {
    const float* x       = (const float*)d_in[0];
    const int*   eidx    = (const int*)d_in[1];
    const float* eattr   = (const float*)d_in[2];
    const float* conv1_W = (const float*)d_in[3];  const float* conv1_b = (const float*)d_in[4];
    const float* e_W1    = (const float*)d_in[5];  const float* e_b1    = (const float*)d_in[6];
    const float* e_W2    = (const float*)d_in[7];  const float* e_b2    = (const float*)d_in[8];
    const float* conv2_W = (const float*)d_in[9];  const float* conv2_b = (const float*)d_in[10];
    const float* conv3_W = (const float*)d_in[11]; const float* conv3_b = (const float*)d_in[12];
    const float* lin1_W  = (const float*)d_in[13]; const float* lin1_b  = (const float*)d_in[14];
    const float* lin2_W  = (const float*)d_in[15]; const float* lin2_b  = (const float*)d_in[16];
    const float* lin3_W  = (const float*)d_in[17]; const float* lin3_b  = (const float*)d_in[18];
    const float* lin4_W  = (const float*)d_in[19]; const float* lin4_b  = (const float*)d_in[20];

    const int N = in_sizes[0] / 64;   // 20000
    const int E = in_sizes[1] / 2;    // 640000
    const int* src = eidx;
    const int* dstp = eidx + E;

    char* ws = (char*)d_ws;
    size_t off = 0;
    auto alloc = [&](size_t bytes) -> void* {
        off = (off + 255) & ~(size_t)255;
        void* p = ws + off;
        off += bytes;
        return p;
    };

    float* dis    = (float*)alloc((size_t)N * 4);
    float* hwbuf  = (float*)alloc((size_t)N * 256 * 4);
    float* accbuf = (float*)alloc((size_t)N * 256 * 4);
    float* out1   = (float*)alloc((size_t)N * 256 * 4);
    float* aggb   = (float*)alloc((size_t)N * 256 * 4);
    float* catb   = (float*)alloc((size_t)N * 512 * 4);
    _Float16* p_conv1 = (_Float16*)alloc(16384 * 2);
    _Float16* p_eW1   = (_Float16*)alloc(49152 * 2);
    _Float16* p_eW2   = (_Float16*)alloc(65536 * 2);
    _Float16* p_conv2 = (_Float16*)alloc(131072 * 2);
    _Float16* p_conv3 = (_Float16*)alloc(16384 * 2);
    _Float16* p_lin1  = (_Float16*)alloc(8192 * 2);
    _Float16* p_lin2  = (_Float16*)alloc(8192 * 2);
    _Float16* p_lin3  = (_Float16*)alloc(4096 * 2);

    auto blocks = [](long n) { return (int)((n + 255) / 256); };

    // 1) repack weights to f16 WMMA fragment layout
    repack_kernel<<<blocks(16384), 256, 0, stream>>>(conv1_W, p_conv1, 64, 256);
    repack_kernel<<<blocks(49152), 256, 0, stream>>>(e_W1,    p_eW1,  192, 256);
    repack_kernel<<<blocks(65536), 256, 0, stream>>>(e_W2,    p_eW2,  256, 256);
    repack_kernel<<<blocks(131072),256, 0, stream>>>(conv2_W, p_conv2,512, 256);
    repack_kernel<<<blocks(16384), 256, 0, stream>>>(conv3_W, p_conv3,256,  64);
    repack_kernel<<<blocks(8192),  256, 0, stream>>>(lin1_W,  p_lin1, 128,  64);
    repack_kernel<<<blocks(8192),  256, 0, stream>>>(lin2_W,  p_lin2,  64, 128);
    repack_kernel<<<blocks(4096),  256, 0, stream>>>(lin3_W,  p_lin3, 128,  32);

    // 2) degree (with self loop) -> dis = rsqrt(deg)
    fill_kernel<<<blocks(N), 256, 0, stream>>>(dis, 1.0f, (long)N);
    deg_kernel<<<blocks(E), 256, 0, stream>>>(dstp, dis, E);
    rsqrt_kernel<<<blocks(N), 256, 0, stream>>>(dis, N);

    auto gemm = [&](const float* A, int lda, const _Float16* Wp, const float* bias,
                    float* C, int ldc, int K, int Ncol, int act) {
        dim3 g(N / 32, (Ncol + 127) / 128);
        size_t shmem = (size_t)32 * K * 2;
        wmma_gemm_kernel<<<g, 256, shmem, stream>>>(A, lda, Wp, bias, C, ldc, K, Ncol, act);
    };

    // 3) conv1 (GCN): hw = x @ W1 ; acc = hw*dis^2 + scatter ; out1 = relu(acc + b)
    gemm(x, 64, p_conv1, nullptr, hwbuf, 256, 64, 256, 0);
    gcn_init_kernel<<<blocks((long)N * 256), 256, 0, stream>>>(hwbuf, dis, accbuf, (long)N * 256, 8);
    gcn_scatter_kernel<<<E, 256, 0, stream>>>(hwbuf, src, dstp, dis, accbuf, E, 256);
    bias_act_kernel<<<blocks((long)N * 256), 256, 0, stream>>>(accbuf, conv1_b, out1, (long)N * 256, 255, 1);

    // 4) EdgeConv with max aggregation
    fill_kernel<<<blocks((long)N * 256), 256, 0, stream>>>(aggb, -INFINITY, (long)N * 256);
    edgeconv_kernel<<<(E + 15) / 16, 256, 0, stream>>>(x, eattr, src, dstp,
                                                       p_eW1, e_b1, p_eW2, e_b2, aggb, E);
    agg_fin_kernel<<<blocks((long)N * 256), 256, 0, stream>>>(aggb, (long)N * 256);

    // 5) conv2 on cat[out1, agg] (N x 512)
    concat_kernel<<<blocks((long)N * 512), 256, 0, stream>>>(out1, 256, aggb, 256, catb, (long)N * 512, 9);
    gemm(catb, 512, p_conv2, nullptr, hwbuf, 256, 512, 256, 0);
    gcn_init_kernel<<<blocks((long)N * 256), 256, 0, stream>>>(hwbuf, dis, accbuf, (long)N * 256, 8);
    gcn_scatter_kernel<<<E, 256, 0, stream>>>(hwbuf, src, dstp, dis, accbuf, E, 256);
    bias_act_kernel<<<blocks((long)N * 256), 256, 0, stream>>>(accbuf, conv2_b, out1, (long)N * 256, 255, 1);  // out2

    // 6) conv3 (256 -> 64), out3 into aggb
    gemm(out1, 256, p_conv3, nullptr, hwbuf, 64, 256, 64, 0);
    gcn_init_kernel<<<blocks((long)N * 64), 256, 0, stream>>>(hwbuf, dis, accbuf, (long)N * 64, 6);
    gcn_scatter_kernel<<<E, 64, 0, stream>>>(hwbuf, src, dstp, dis, accbuf, E, 64);
    bias_act_kernel<<<blocks((long)N * 64), 256, 0, stream>>>(accbuf, conv3_b, aggb, (long)N * 64, 63, 1);  // out3

    // 7) MLP head: cat[x, out3] -> softplus chain -> [N,2]
    concat_kernel<<<blocks((long)N * 128), 256, 0, stream>>>(x, 64, aggb, 64, catb, (long)N * 128, 7);
    gemm(catb, 128, p_lin1, lin1_b, hwbuf, 64, 128, 64, 2);
    gemm(hwbuf, 64, p_lin2, lin2_b, accbuf, 128, 64, 128, 2);
    gemm(accbuf, 128, p_lin3, lin3_b, out1, 32, 128, 32, 2);
    final_kernel<<<blocks(N), 256, 0, stream>>>(out1, lin4_W, lin4_b, (float*)d_out, N);
}
